// MultiScaleDeformableAttention_45930380263459
// MI455X (gfx1250) — matrix operand order repaired and checked by reading, not compiled, not run on hardware
//
#include <hip/hip_runtime.h>
#include <hip/hip_bf16.h>

// ---------------------------------------------------------------------------
// MultiScaleDeformableAttention for MI455X (gfx1250, wave32, WMMA)
//
//  1. value = input_flatten @ Wv + bv        (WMMA bf16 GEMM, M=87040,N=256)
//  2. off   = query @ Wo + bo                (WMMA bf16 GEMM, M=16000,N=256)
//  3. attnl = query @ Wa + ba                (WMMA bf16 GEMM, M=16000,N=128)
//  4. sampling: softmax + bilinear gather    (wave per (b,q,h), lane = channel)
//  5. out   = mid @ Wout + bout              (WMMA bf16 GEMM, M=16000,N=256)
//
// All phases are memory-bound (~300 MB streamed at 23.3 TB/s ≈ 13 us; the
// 89 MB value table fits in the 192 MB L2 so the gather phase rides L2).
// N is a template parameter (256 / 128) so the whole epilogue collapses to
// immediate-offset stores off a single 64-bit base.
// ---------------------------------------------------------------------------

#define BATCH 16
#define QLEN  1000
#define DMODEL 256
#define HEADS 8
#define DHEAD 32
#define LVLS  4
#define TPTS  16
#define LTOT  5440
#define KDIM  256

// padded LDS strides (words): conflict-spread while keeping 16B alignment
#define ASTRIDE 132   // 128 kpairs + 4 pad  (132 % 64 == 4)
#define BSTRIDE 20    // 16 kpairs + 4 pad   (20*col hits all 4-word groups)

typedef __attribute__((ext_vector_type(16))) __bf16 v16bf;
typedef __attribute__((ext_vector_type(8)))  float  v8f;

union FragAB {
    unsigned u[8];
    v16bf    v;
};

static __device__ __forceinline__ unsigned short f2bf(float f) {
    unsigned u = __float_as_uint(f);
    unsigned r = (u + 0x7FFFu + ((u >> 16) & 1u)) >> 16;   // round-to-nearest-even
    return (unsigned short)r;
}
static __device__ __forceinline__ unsigned pack2bf(float lo, float hi) {
    return (unsigned)f2bf(lo) | ((unsigned)f2bf(hi) << 16);
}

// ---------------------------------------------------------------------------
// GEMM with bias:  C[M,NDIM] = A[M,256] @ Bw[256,NDIM] + bias[NDIM]
// fp32 in / fp32 out, bf16 WMMA, fp32 accumulation.
// Block: 512 threads = 16 waves; tile 64(M) x 256(N).
// M is a multiple of 64 at every call site (no edge guards).
// ---------------------------------------------------------------------------
template <int NDIM>
__global__ __launch_bounds__(512) void gemm_bias_wmma(
    const float* __restrict__ A, const float* __restrict__ Bw,
    const float* __restrict__ bias, float* __restrict__ C,
    int M)
{
    __shared__ unsigned A_pack[64 * ASTRIDE];        // ~33.8 KB
    __shared__ unsigned B_pack[2][256 * BSTRIDE];    // 2 x 20 KB (double buffer)

    const int tid  = threadIdx.x;
    const int row0 = blockIdx.x * 64;

    // warm L2 for the next block's A tile: 512 threads x 128 B = one 64 KB tile
    if (row0 + 64 < M)
        __builtin_prefetch(A + (size_t)(row0 + 64) * KDIM + tid * 32, 0, 1);

    // ---- stage A tile (once): fp32 -> packed bf16x2, [m][kpair] ----
    for (int i = tid; i < 64 * 128; i += 512) {
        int m  = i >> 7;
        int kp = i & 127;
        float2 f = ((const float2*)(A + (size_t)(row0 + m) * KDIM))[kp];
        A_pack[m * ASTRIDE + kp] = pack2bf(f.x, f.y);
    }

    // ---- B panel fill: n fast -> coalesced row reads of Bw ----
    auto fillB = [&](int buf, int kb) {
#pragma unroll 1
        for (int i = tid; i < NDIM * 16; i += 512) {
            int n  = i & (NDIM - 1);
            int kp = i / NDIM;             // 0..15 (compile-time shift)
            int k  = kb + kp * 2;
            float b0 = Bw[(size_t)k * NDIM + n];
            float b1 = Bw[(size_t)(k + 1) * NDIM + n];
            B_pack[buf][n * BSTRIDE + kp] = pack2bf(b0, b1);
        }
    };

    fillB(0, 0);

    const int wave  = tid >> 5;
    const int lane  = tid & 31;
    const int ntile = wave;                    // wave w owns columns [16w, 16w+16)
    const int nl    = lane & 15;
    const int hh    = lane >> 4;               // lane half-group (ISA layout)
    const bool active = (ntile * 16) < NDIM;   // wave-uniform -> EXEC all-ones

    v8f acc[4];
#pragma unroll
    for (int ms = 0; ms < 4; ++ms)
        acc[ms] = (v8f){0.f, 0.f, 0.f, 0.f, 0.f, 0.f, 0.f, 0.f};

    __syncthreads();

    for (int kb = 0; kb < KDIM; kb += 32) {
        const int cur = (kb >> 5) & 1;

        // issue the next panel's global loads before this step's WMMAs
        if (kb + 32 < KDIM)
            fillB(cur ^ 1, kb + 32);

        if (active) {
            // B fragment (32x16 KxN): VGPR j holds pair k0 = (j>>2)*16 + hh*8 + (j&3)*2 at col nl
            FragAB bf;
            const unsigned* bp = &B_pack[cur][(ntile * 16 + nl) * BSTRIDE + hh * 4];
#pragma unroll
            for (int j = 0; j < 8; ++j)
                bf.u[j] = bp[(j >> 2) * 8 + (j & 3)];

            const int kb2 = kb >> 1;
#pragma unroll
            for (int ms = 0; ms < 4; ++ms) {
                // A fragment (16x32): VGPR j holds pair k0 = (j>>2)*16 + hh*8 + (j&3)*2
                FragAB af;
                const unsigned* ap = &A_pack[(ms * 16 + nl) * ASTRIDE + kb2 + hh * 4];
#pragma unroll
                for (int j = 0; j < 8; ++j)
                    af.u[j] = ap[(j >> 2) * 8 + (j & 3)];

                acc[ms] = __builtin_amdgcn_wmma_f32_16x16x32_bf16(
                    /*neg_a=*/false, af.v, /*neg_b=*/false, bf.v,
                    /*c_mod=*/(short)0, acc[ms],
                    /*reuse_a=*/false, /*reuse_b=*/false);
            }
        }

        __syncthreads();
    }

    if (active) {
        const int col  = ntile * 16 + nl;
        const float bv = bias[col];
        // C/D layout: lanes 0-15 -> M=r, lanes 16-31 -> M=8+r; N = lane&15
        // Single 64-bit base; all row offsets are compile-time immediates.
        float* crow = C + (size_t)(row0 + 8 * hh) * NDIM + col;
#pragma unroll
        for (int ms = 0; ms < 4; ++ms) {
#pragma unroll
            for (int r = 0; r < 8; ++r)
                crow[(ms * 16 + r) * NDIM] = acc[ms][r] + bv;
        }
    }
}

// ---------------------------------------------------------------------------
// Sampling: one wave per (b,q,h); lane = channel within head (DHEAD == 32).
// ---------------------------------------------------------------------------
__global__ __launch_bounds__(256) void msda_sample(
    const float* __restrict__ refpts,   // [B,Q,LV,2]
    const float* __restrict__ value,    // [B,LTOT,256] (h*32+dh channel)
    const float* __restrict__ offb,     // [B*Q,256] = (h, lv, np, 2)
    const float* __restrict__ attnb,    // [B*Q,128] = (h, lv, np)
    float* __restrict__ mid)            // [B*Q,256]
{
    const int wid  = blockIdx.x * 8 + (threadIdx.x >> 5);
    const int lane = threadIdx.x & 31;

    const int h  = wid & 7;
    const int bq = wid >> 3;             // b*1000 + q
    const int b  = bq / QLEN;

    const int HlA[4]    = {64, 32, 16, 8};
    const int WlA[4]    = {64, 32, 16, 8};
    const int startA[4] = {0, 4096, 5120, 5376};

    // softmax over the 16 logits (same-address loads across lanes -> broadcast)
    float lg[TPTS];
    float mx = -3.0e38f;
#pragma unroll
    for (int p = 0; p < TPTS; ++p) {
        lg[p] = attnb[(size_t)bq * 128 + h * 16 + p];
        mx = fmaxf(mx, lg[p]);
    }
    float s = 0.f;
#pragma unroll
    for (int p = 0; p < TPTS; ++p) {
        lg[p] = __expf(lg[p] - mx);
        s += lg[p];
    }
    const float inv = 1.f / s;

    const float* vb = value + ((size_t)b * LTOT) * DMODEL + h * DHEAD + lane;

    float acc = 0.f;
#pragma unroll
    for (int p = 0; p < TPTS; ++p) {
        const int lv = p >> 2;
        const int Wl = WlA[lv], Hl = HlA[lv];

        const float rx = refpts[((size_t)bq * LVLS + lv) * 2 + 0];
        const float ry = refpts[((size_t)bq * LVLS + lv) * 2 + 1];
        const float ox = offb[(size_t)bq * 256 + h * 32 + p * 2 + 0];
        const float oy = offb[(size_t)bq * 256 + h * 32 + p * 2 + 1];

        // loc = ref + off/norm; pixel = loc*size - 0.5  (align_corners=False)
        const float x = rx * (float)Wl + ox - 0.5f;
        const float y = ry * (float)Hl + oy - 0.5f;

        const float x0f = floorf(x), y0f = floorf(y);
        const float lx = x - x0f, ly = y - y0f;
        const int x0 = (int)x0f, y0 = (int)y0f;

        const float aw = lg[p] * inv;
        const float wx[2] = {1.f - lx, lx};
        const float wy[2] = {1.f - ly, ly};

#pragma unroll
        for (int c = 0; c < 4; ++c) {
            const int ix = x0 + (c & 1);
            const int iy = y0 + (c >> 1);
            if (ix >= 0 && ix < Wl && iy >= 0 && iy < Hl) {
                const float w = wx[c & 1] * wy[c >> 1];
                acc += aw * w * vb[(size_t)(startA[lv] + iy * Wl + ix) * DMODEL];
            }
        }
    }

    mid[(size_t)bq * DMODEL + h * DHEAD + lane] = acc;
}

// ---------------------------------------------------------------------------
// Launch
// ---------------------------------------------------------------------------
extern "C" void kernel_launch(void* const* d_in, const int* in_sizes, int n_in,
                              void* d_out, int out_size, void* d_ws, size_t ws_size,
                              hipStream_t stream) {
    const float* query  = (const float*)d_in[0];
    const float* refpts = (const float*)d_in[1];
    const float* inflat = (const float*)d_in[2];
    // d_in[3] (shapes) / d_in[4] (level starts) are static -> hard-coded
    const float* Wv   = (const float*)d_in[5];
    const float* bv   = (const float*)d_in[6];
    const float* Wo   = (const float*)d_in[7];
    const float* bo   = (const float*)d_in[8];
    const float* Wa   = (const float*)d_in[9];
    const float* ba   = (const float*)d_in[10];
    const float* Wout = (const float*)d_in[11];
    const float* bout = (const float*)d_in[12];

    float* ws    = (float*)d_ws;
    float* value = ws;                                    // 16*5440*256
    float* offb  = value + (size_t)BATCH * LTOT * DMODEL; // 16000*256
    float* attnb = offb + (size_t)BATCH * QLEN * 256;     // 16000*128
    float* mid   = attnb + (size_t)BATCH * QLEN * 128;    // 16000*256
    float* out   = (float*)d_out;

    const int MV = BATCH * LTOT;   // 87040
    const int MQ = BATCH * QLEN;   // 16000

    gemm_bias_wmma<256><<<MV / 64, 512, 0, stream>>>(inflat, Wv, bv, value, MV);
    gemm_bias_wmma<256><<<MQ / 64, 512, 0, stream>>>(query, Wo, bo, offb, MQ);
    gemm_bias_wmma<128><<<MQ / 64, 512, 0, stream>>>(query, Wa, ba, attnb, MQ);
    msda_sample<<<(BATCH * QLEN * HEADS) / 8, 256, 0, stream>>>(refpts, value, offb, attnb, mid);
    gemm_bias_wmma<256><<<MQ / 64, 512, 0, stream>>>(mid, Wout, bout, out, MQ);
}
